// HGCN_83597243449448
// MI455X (gfx1250) — matrix operand order repaired and checked by reading, not compiled, not run on hardware
//
#include <hip/hip_runtime.h>
#include <hip/hip_bf16.h>

// ---------------------------------------------------------------------------
// HGCN forward for MI455X (gfx1250, wave32, WMMA + TDM).
// Pipeline: deg-norms -> conv1 (WMMA GEMM + atomic scatter) -> L2 norm ->
//           3-head GAT (WMMA GEMM + 3-pass edge softmax) -> relu ->
//           conv2 per head (WMMA GEMM + scatter) -> sigmoid -> mean pool ->
//           classifier.
// GEMMs: activation tile staged to LDS by the Tensor Data Mover
// (tensor_load_to_lds, TENSORcnt), weights staged to LDS as transposed f16,
// K fully unrolled -> K/32 v_wmma_f32_16x16x32_f16 per wave.
// ---------------------------------------------------------------------------

typedef __attribute__((ext_vector_type(16))) _Float16 v16h;
typedef __attribute__((ext_vector_type(8)))  float    v8f;
typedef __attribute__((ext_vector_type(4)))  unsigned int v4u;
typedef __attribute__((ext_vector_type(8)))  int      v8i;
typedef __attribute__((ext_vector_type(4)))  int      v4i;

#define RREL   3
#define NHEAD  3
#define FHID   64
#define FIN    128
#define FOUT   64
#define NCLS   16
#define NGRAPH 8
#define GATF   (NHEAD * FHID)   // 192

// ---------------- utility kernels ----------------

__global__ void fill_kernel(float* __restrict__ p, float v, int n) {
  int i = blockIdx.x * blockDim.x + threadIdx.x;
  int stride = gridDim.x * blockDim.x;
  for (; i < n; i += stride) p[i] = v;
}

__global__ void degree_kernel(const int* __restrict__ idx, float* __restrict__ deg, int E) {
  int i = blockIdx.x * blockDim.x + threadIdx.x;
  int stride = gridDim.x * blockDim.x;
  for (; i < E; i += stride) atomicAdd(&deg[idx[i]], 1.0f);
}

__global__ void norm_kernel(float* __restrict__ deg, int n) {
  int i = blockIdx.x * blockDim.x + threadIdx.x;
  int stride = gridDim.x * blockDim.x;
  for (; i < n; i += stride) deg[i] = rsqrtf(fmaxf(deg[i], 1.0f));
}

__device__ inline void atomicMaxF(float* addr, float val) {
  int* ai = reinterpret_cast<int*>(addr);
  int old = __float_as_int(*addr);
  while (__int_as_float(old) < val) {
    int assumed = old;
    old = atomicCAS(ai, assumed, __float_as_int(val));
    if (old == assumed) break;
  }
}

// ---------------- WMMA GEMM with TDM-staged activations ----------------
// Y[M,NOUT] = diag(rowscale) * X[M,K] @ W[K,NOUT]
// grid = M/16 blocks; blockDim = (NOUT/16)*32; wave w computes cols [16w,16w+16).
// X 16-row stripe -> LDS via tensor_load_to_lds (f32, D# per CDNA5 ISA ch.8);
// W -> LDS transposed f16 (per-lane B fragment halves contiguous).
// Fragment layouts used by this code (wave32; A and B share contiguous k):
//   A 16x32 f16 : lane L -> row M=L&15; kbase 0 (L<16) / 16 (L>=16);
//                 half i -> K=kbase+i
//   B 32x16 f16 : lane L -> col N=L&15; kbase 0 (L<16) / 16 (L>=16); half i -> K=kbase+i
//   C/D 16x16 f32: VGPR v -> M = v + (L<16 ? 0 : 8), N = L&15
template <int K, int NOUT>
__global__ void gemm_wmma(const float* __restrict__ X, const float* __restrict__ W,
                          const float* __restrict__ rowscale, int scaleDiv,
                          float* __restrict__ Y, int M) {
  __shared__ float    sX[16 * K];        // activation stripe (f32, filled by TDM)
  __shared__ _Float16 sWt[NOUT * K];     // weights, transposed: sWt[n*K + k]

  const int lane = threadIdx.x & 31;
  const int wave = threadIdx.x >> 5;     // == column tile index
  const int row0 = blockIdx.x << 4;

  // --- wave 0: issue TDM load of X[row0:row0+16, 0:K] into sX ---
  if (wave == 0) {
    const unsigned long long ga =
        (unsigned long long)(uintptr_t)(X + (size_t)row0 * K);
    const unsigned int ldsa = (unsigned int)(uintptr_t)sX;
    v4u g0;
    g0[0] = 1u;                                            // count=1 valid descriptor
    g0[1] = ldsa;                                          // lds_addr [63:32]
    g0[2] = (unsigned int)(ga & 0xFFFFFFFFull);            // global_addr low
    g0[3] = (unsigned int)((ga >> 32) & 0x01FFFFFFull)     // global_addr [56:32]
            | (2u << 30);                                  // type = 2 ("image")
    v8i g1;
    g1[0] = (int)(2u << 16);                               // data_size = 4 bytes
    g1[1] = (int)(((unsigned)K & 0xFFFFu) << 16);          // tensor_dim0[15:0]
    g1[2] = (int)((((unsigned)K >> 16) & 0xFFFFu)          // tensor_dim0[31:16]
            | (((unsigned)M & 0xFFFFu) << 16));            // tensor_dim1[15:0]
    g1[3] = (int)((((unsigned)M >> 16) & 0xFFFFu)          // tensor_dim1[31:16]
            | (((unsigned)K & 0xFFFFu) << 16));            // tile_dim0 = K
    g1[4] = 16;                                            // tile_dim1 = 16 rows
    g1[5] = K;                                             // tensor_dim0_stride low32
    g1[6] = 0;
    g1[7] = 0;
    v4i z4 = {0, 0, 0, 0};
    v8i z8 = {0, 0, 0, 0, 0, 0, 0, 0};
    __builtin_amdgcn_tensor_load_to_lds(g0, g1, z4, z4, z8, 0);  // 6-arg (clang-23) form
    __builtin_amdgcn_s_wait_tensorcnt(0);                  // s_wait_tensorcnt 0
  }

  // --- all waves: stage W transposed as f16 (coalesced global reads) ---
  for (int i = threadIdx.x; i < K * NOUT; i += blockDim.x) {
    const int k = i / NOUT, n = i - k * NOUT;
    sWt[n * K + k] = (_Float16)W[i];
  }
  __syncthreads();

  const int lr     = lane & 15;                  // local row / local col
  const int rowA   = row0 + lr;
  const float sc   = rowscale ? rowscale[rowA / scaleDiv] : 1.0f;
  const int kbaseA = (lane < 16) ? 0 : 16;
  const int kbaseB = (lane < 16) ? 0 : 16;
  const _Float16* wcol = &sWt[(wave * 16 + lr) * K];
  const float*    xrow = &sX[lr * K];

  v8f acc = {0.f, 0.f, 0.f, 0.f, 0.f, 0.f, 0.f, 0.f};
#pragma unroll
  for (int kb = 0; kb < K; kb += 32) {
    v16h a, b;
#pragma unroll
    for (int i = 0; i < 16; ++i) {
      const int vg = i >> 1, w = i & 1;
      const int ka = kbaseA + 2 * vg + w;
      a[i] = (_Float16)(xrow[kb + ka] * sc);
      b[i] = wcol[kb + kbaseB + i];              // 16 contiguous halves -> wide ds loads
    }
    acc = __builtin_amdgcn_wmma_f32_16x16x32_f16(false, a, false, b, (short)0, acc,
                                                 false, false);
  }

  const int mBase = row0 + ((lane < 16) ? 0 : 8);
  const int nOut  = wave * 16 + lr;
#pragma unroll
  for (int v = 0; v < 8; ++v)
    Y[(size_t)(mBase + v) * NOUT + nOut] = acc[v];
}

// ---------------- edge scatter: acc[d,:] += nd[d] * hw[s,:] ----------------
__global__ void scatter_kernel(const float* __restrict__ hw, const int* __restrict__ src,
                               const int* __restrict__ dst, const float* __restrict__ nd,
                               float* __restrict__ acc, int E, int F) {
  const int c = threadIdx.x;  // blockDim.x == F
  for (int e = blockIdx.x; e < E; e += gridDim.x) {
    const int s = src[e], d = dst[e];
    const int en = e + gridDim.x;
    if (en < E && (c & 63) == 0)
      __builtin_prefetch(&hw[(size_t)src[en] * F + c], 0, 0);  // global_prefetch_b8
    atomicAdd(&acc[(size_t)d * F + c], nd[d] * hw[(size_t)s * F + c]);
  }
}

// ---------------- conv1 epilogue: h += sum_r b1[r]; L2 normalize rows ----------------
__global__ void bias_l2norm_kernel(float* __restrict__ h1, const float* __restrict__ b1, int N) {
  __shared__ float s[FHID];
  const int t = threadIdx.x;
  for (int n = blockIdx.x; n < N; n += gridDim.x) {
    float v = h1[(size_t)n * FHID + t] + b1[t] + b1[FHID + t] + b1[2 * FHID + t];
    s[t] = v * v;
    __syncthreads();
    for (int off = 32; off > 0; off >>= 1) {
      if (t < off) s[t] += s[t + off];
      __syncthreads();
    }
    const float inv = 1.0f / fmaxf(sqrtf(s[0]), 1e-12f);
    h1[(size_t)n * FHID + t] = v * inv;
    __syncthreads();
  }
}

// ---------------- GAT: per-node attention logits el/er ----------------
__global__ void el_er_kernel(const float* __restrict__ f, const float* __restrict__ al,
                             const float* __restrict__ ar, float* __restrict__ el,
                             float* __restrict__ er, int N) {
  __shared__ float s1[FHID], s2[FHID];
  const int t = threadIdx.x;
  for (int n = blockIdx.x; n < N; n += gridDim.x) {
    for (int h = 0; h < NHEAD; ++h) {
      const float fv = f[(size_t)n * GATF + h * FHID + t];
      s1[t] = fv * al[h * FHID + t];
      s2[t] = fv * ar[h * FHID + t];
      __syncthreads();
      for (int off = 32; off > 0; off >>= 1) {
        if (t < off) { s1[t] += s1[t + off]; s2[t] += s2[t + off]; }
        __syncthreads();
      }
      if (t == 0) { el[n * NHEAD + h] = s1[0]; er[n * NHEAD + h] = s2[0]; }
      __syncthreads();
    }
  }
}

__device__ inline float edge_logit(const float* el, const float* er, int s, int d, int h) {
  float ev = el[s * NHEAD + h] + er[d * NHEAD + h];
  return ev > 0.f ? ev : 0.2f * ev;  // leaky_relu(0.2)
}

__global__ void gat_max_kernel(const int* __restrict__ src, const int* __restrict__ dst,
                               const float* __restrict__ el, const float* __restrict__ er,
                               float* __restrict__ mx, int total) {
  int i = blockIdx.x * blockDim.x + threadIdx.x;
  const int stride = gridDim.x * blockDim.x;
  for (; i < total; i += stride) {
    const int h = i % NHEAD, e = i / NHEAD;
    const int s = src[e], d = dst[e];
    atomicMaxF(&mx[d * NHEAD + h], edge_logit(el, er, s, d, h));
  }
}

__global__ void gat_sum_kernel(const int* __restrict__ src, const int* __restrict__ dst,
                               const float* __restrict__ el, const float* __restrict__ er,
                               const float* __restrict__ mx, float* __restrict__ zz, int total) {
  int i = blockIdx.x * blockDim.x + threadIdx.x;
  const int stride = gridDim.x * blockDim.x;
  for (; i < total; i += stride) {
    const int h = i % NHEAD, e = i / NHEAD;
    const int s = src[e], d = dst[e];
    atomicAdd(&zz[d * NHEAD + h], __expf(edge_logit(el, er, s, d, h) - mx[d * NHEAD + h]));
  }
}

__global__ void gat_acc_kernel(const int* __restrict__ src, const int* __restrict__ dst,
                               const float* __restrict__ el, const float* __restrict__ er,
                               const float* __restrict__ mx, const float* __restrict__ zz,
                               const float* __restrict__ f, float* __restrict__ h2, int E) {
  const int c = threadIdx.x;        // blockDim.x == GATF
  const int h = c / FHID;
  for (int e = blockIdx.x; e < E; e += gridDim.x) {
    const int s = src[e], d = dst[e];
    const float ez = __expf(edge_logit(el, er, s, d, h) - mx[d * NHEAD + h]);
    const float alpha = ez / (zz[d * NHEAD + h] + 1e-9f);
    atomicAdd(&h2[(size_t)d * GATF + c], alpha * f[(size_t)s * GATF + c]);
  }
}

__global__ void bias_relu_kernel(float* __restrict__ h2, const float* __restrict__ bg, int total) {
  int i = blockIdx.x * blockDim.x + threadIdx.x;
  const int stride = gridDim.x * blockDim.x;
  for (; i < total; i += stride) {
    const int c = i % GATF;
    const float v = h2[i] + bg[c] + bg[GATF + c] + bg[2 * GATF + c];
    h2[i] = v > 0.f ? v : 0.f;
  }
}

// ---------------- sigmoid + mean-node pooling ----------------
__global__ void sigmoid_pool_kernel(const float* __restrict__ h3, const float* __restrict__ b2,
                                    const int* __restrict__ gid, float* __restrict__ pooled,
                                    float* __restrict__ cnt, int N) {
  const int c = threadIdx.x;  // blockDim.x == GATF
  const int co = c % FOUT;
  const float bsum = b2[co] + b2[FOUT + co] + b2[2 * FOUT + co];
  for (int n = blockIdx.x; n < N; n += gridDim.x) {
    const int g = gid[n];
    const float v = 1.0f / (1.0f + __expf(-(h3[(size_t)n * GATF + c] + bsum)));
    atomicAdd(&pooled[g * GATF + c], v);
    if (c == 0) atomicAdd(&cnt[g], 1.0f);
  }
}

// ---------------- classifier: out[b,h] = mean_c sigmoid(pooled_mean @ Wc + bc) ----------------
__global__ void final_kernel(const float* __restrict__ pooled, const float* __restrict__ cnt,
                             const float* __restrict__ Wc, const float* __restrict__ bc,
                             float* __restrict__ out) {
  __shared__ float s[NGRAPH * NHEAD * NCLS];
  const int t = threadIdx.x;                 // 0..383
  const int b = t / (NHEAD * NCLS);
  const int h = (t / NCLS) % NHEAD;
  const int c = t % NCLS;
  const float invc = 1.0f / fmaxf(cnt[b], 1.0f);
  float acc = bc[c];
  for (int o = 0; o < FOUT; ++o)
    acc += pooled[b * GATF + h * FOUT + o] * invc * Wc[o * NCLS + c];
  s[t] = 1.0f / (1.0f + __expf(-acc));
  __syncthreads();
  if (c == 0) {
    float m = 0.f;
    for (int i = 0; i < NCLS; ++i) m += s[t + i];
    out[b * NHEAD + h] = m / (float)NCLS;
  }
}

// ---------------------------------------------------------------------------

extern "C" void kernel_launch(void* const* d_in, const int* in_sizes, int n_in,
                              void* d_out, int out_size, void* d_ws, size_t ws_size,
                              hipStream_t stream) {
  const float* x   = (const float*)d_in[0];
  const int*   src = (const int*)  d_in[1];
  const int*   dst = (const int*)  d_in[2];
  const int*   gid = (const int*)  d_in[3];
  const float* W1  = (const float*)d_in[4];
  const float* b1  = (const float*)d_in[5];
  const float* Wg  = (const float*)d_in[6];
  const float* al  = (const float*)d_in[7];
  const float* ar  = (const float*)d_in[8];
  const float* bg  = (const float*)d_in[9];
  const float* W2  = (const float*)d_in[10];
  const float* b2  = (const float*)d_in[11];
  const float* Wc  = (const float*)d_in[12];
  const float* bc  = (const float*)d_in[13];

  const int N = in_sizes[0] / FIN;     // 50000 (divisible by 16; N*3 divisible by 16)
  const int E = in_sizes[1] / RREL;    // 400000

  // workspace layout (floats)
  float* ws = (float*)d_ws;
  size_t o = 0;
  float* ns  = ws + o; o += (size_t)RREL * N;     // src-degree norms per relation
  float* nd  = ws + o; o += (size_t)RREL * N;     // dst-degree norms per relation
  float* h1  = ws + o; o += (size_t)N * FHID;     // conv1 output
  float* tmp = ws + o; o += (size_t)N * GATF;     // hw / GAT feature buffer (reused per r)
  float* el  = ws + o; o += (size_t)N * NHEAD;
  float* er  = ws + o; o += (size_t)N * NHEAD;
  float* mx  = ws + o; o += (size_t)N * NHEAD;
  float* zz  = ws + o; o += (size_t)N * NHEAD;
  float* h2  = ws + o; o += (size_t)N * GATF;     // GAT output
  float* h3  = ws + o; o += (size_t)N * GATF;     // conv2 output
  float* pooled = ws + o; o += (size_t)NGRAPH * GATF;
  float* cnt = ws + o; o += NGRAPH;
  if (o * sizeof(float) > ws_size) return;

  auto cdiv = [](int a, int b) { return (a + b - 1) / b; };

  // ---- degree norms for all 6 (relation, role) index arrays ----
  fill_kernel<<<cdiv(2 * RREL * N, 256), 256, 0, stream>>>(ns, 0.f, 2 * RREL * N);
  for (int r = 0; r < RREL; ++r) {
    degree_kernel<<<cdiv(E, 256), 256, 0, stream>>>(src + (size_t)r * E, ns + (size_t)r * N, E);
    degree_kernel<<<cdiv(E, 256), 256, 0, stream>>>(dst + (size_t)r * E, nd + (size_t)r * N, E);
  }
  norm_kernel<<<cdiv(2 * RREL * N, 256), 256, 0, stream>>>(ns, 2 * RREL * N);

  // ---- conv1: h1 = sum_r scatter(nd_r * ((x*ns_r) @ W1_r)) ----
  fill_kernel<<<cdiv(N * FHID, 256), 256, 0, stream>>>(h1, 0.f, N * FHID);
  for (int r = 0; r < RREL; ++r) {
    gemm_wmma<FIN, FHID><<<N / 16, (FHID / 16) * 32, 0, stream>>>(
        x, W1 + (size_t)r * FIN * FHID, ns + (size_t)r * N, 1, tmp, N);
    scatter_kernel<<<E, FHID, 0, stream>>>(tmp, src + (size_t)r * E, dst + (size_t)r * E,
                                           nd + (size_t)r * N, h1, E, FHID);
  }
  bias_l2norm_kernel<<<N, FHID, 0, stream>>>(h1, b1, N);

  // ---- GAT: h2 = sum_r edge-softmax attention over relation r ----
  fill_kernel<<<cdiv(N * GATF, 256), 256, 0, stream>>>(h2, 0.f, N * GATF);
  for (int r = 0; r < RREL; ++r) {
    gemm_wmma<FHID, GATF><<<N / 16, (GATF / 16) * 32, 0, stream>>>(
        h1, Wg + (size_t)r * FHID * GATF, nullptr, 1, tmp, N);
    el_er_kernel<<<N, FHID, 0, stream>>>(tmp, al + (size_t)r * GATF, ar + (size_t)r * GATF,
                                         el, er, N);
    fill_kernel<<<cdiv(N * NHEAD, 256), 256, 0, stream>>>(mx, -1e30f, N * NHEAD);
    fill_kernel<<<cdiv(N * NHEAD, 256), 256, 0, stream>>>(zz, 0.f, N * NHEAD);
    gat_max_kernel<<<cdiv(E * NHEAD, 256), 256, 0, stream>>>(src + (size_t)r * E, dst + (size_t)r * E,
                                                             el, er, mx, E * NHEAD);
    gat_sum_kernel<<<cdiv(E * NHEAD, 256), 256, 0, stream>>>(src + (size_t)r * E, dst + (size_t)r * E,
                                                             el, er, mx, zz, E * NHEAD);
    gat_acc_kernel<<<E, GATF, 0, stream>>>(src + (size_t)r * E, dst + (size_t)r * E,
                                           el, er, mx, zz, tmp, h2, E);
  }
  bias_relu_kernel<<<cdiv(N * GATF, 256), 256, 0, stream>>>(h2, bg, N * GATF);

  // ---- conv2 (per head): h3 = sum_r scatter(nd_r * ((h2*ns_r) @ W2_r)) ----
  fill_kernel<<<cdiv(N * GATF, 256), 256, 0, stream>>>(h3, 0.f, N * GATF);
  for (int r = 0; r < RREL; ++r) {
    const int M = N * NHEAD;  // rows = (node, head); rowscale indexed by node = row/3
    gemm_wmma<FHID, FOUT><<<M / 16, (FOUT / 16) * 32, 0, stream>>>(
        h2, W2 + (size_t)r * FHID * FOUT, ns + (size_t)r * N, NHEAD, tmp, M);
    scatter_kernel<<<E, GATF, 0, stream>>>(tmp, src + (size_t)r * E, dst + (size_t)r * E,
                                           nd + (size_t)r * N, h3, E, GATF);
  }

  // ---- sigmoid + mean pooling + classifier ----
  fill_kernel<<<cdiv(NGRAPH * GATF + NGRAPH, 256), 256, 0, stream>>>(pooled, 0.f,
                                                                     NGRAPH * GATF + NGRAPH);
  sigmoid_pool_kernel<<<N, GATF, 0, stream>>>(h3, b2, gid, pooled, cnt, N);
  final_kernel<<<1, NGRAPH * NHEAD * NCLS, 0, stream>>>(pooled, cnt, Wc, bc, (float*)d_out);
}
